// SPARTA_56470230008479
// MI455X (gfx1250) — matrix operand order repaired and checked by reading, not compiled, not run on hardware
//
#include <hip/hip_runtime.h>
#include <hip/hip_bf16.h>
#include <math.h>

// ColBERT late-interaction scoring on MI455X (gfx1250, wave32, WMMA).
// out[q][p] = sum_a log1p( relu( max_b <Qe[q,a,:], Pe[p,b,:]> ) )
//
// Compute-bound (AI ~ 505 FLOP/B): use V_WMMA_F32_16X16X32_F16 with fp32
// accumulation; fp32 inputs are converted to f16 while staging into LDS.

typedef __attribute__((ext_vector_type(8)))  _Float16 v8h;
typedef __attribute__((ext_vector_type(16))) _Float16 v16h;
typedef __attribute__((ext_vector_type(8)))  float    v8f;

#define Q_NUM   32
#define LQ_DIM  32
#define P_NUM   256
#define LP_DIM  300
#define D_DIM   768

#define KC      64                 // K-chunk staged per iteration
#define NKC     (D_DIM / KC)       // 12 chunks
#define N_PAD   320                // LP padded to 20 N-tiles (zero cols => free ReLU)
#define A_STR   (KC + 8)           // LDS row strides padded to dodge bank conflicts
#define B_STR   (KC + 8)
#define NTHREADS 320
#define NWAVES  10                 // 10 wave32s; each owns 2 Mtiles x 2 Ntiles

__global__ __launch_bounds__(NTHREADS)
void SPARTA_colbert_wmma(const float* __restrict__ Qe,
                         const float* __restrict__ Pe,
                         float* __restrict__ Out) {
  __shared__ _Float16 As[LQ_DIM * A_STR];    //  4.6 KB
  __shared__ _Float16 Bs[N_PAD * B_STR];     // 46.1 KB
  __shared__ float    WMax[NWAVES * 32];     //  1.3 KB

  const int q      = blockIdx.x;             // consecutive blocks share p => L2 reuse
  const int p      = blockIdx.y;
  const int tid    = threadIdx.x;
  const int wave   = tid >> 5;
  const int lane   = tid & 31;
  const int lane16 = lane & 15;
  const int laneHi = lane >> 4;              // 0: lanes 0-15, 1: lanes 16-31

  const float* Aq = Qe + (size_t)q * LQ_DIM * D_DIM;
  const float* Bp = Pe + (size_t)p * LP_DIM * D_DIM;

  v8f acc[2][2] = {};                        // 2 Mtiles x 2 Ntiles, f32 accum

  for (int kc = 0; kc < NKC; ++kc) {
    const int d0 = kc * KC;
    __syncthreads();                         // previous chunk's compute done

    // ---- stage A chunk: 32 rows x 64 dims, fp32 -> f16 ----
    for (int i = tid; i < LQ_DIM * (KC / 4); i += NTHREADS) {
      const int row = i / (KC / 4);
      const int kv  = i % (KC / 4);
      const float4 f = *(const float4*)(Aq + row * D_DIM + d0 + kv * 4);
      _Float16* dst = &As[row * A_STR + kv * 4];
      dst[0] = (_Float16)f.x; dst[1] = (_Float16)f.y;
      dst[2] = (_Float16)f.z; dst[3] = (_Float16)f.w;
    }
    // ---- stage B chunk: 320 cols x 64 dims (cols >= 300 zeroed) ----
    for (int i = tid; i < N_PAD * (KC / 4); i += NTHREADS) {
      const int col = i / (KC / 4);
      const int kv  = i % (KC / 4);
      _Float16* dst = &Bs[col * B_STR + kv * 4];
      if (col < LP_DIM) {
        const float* src = Bp + col * D_DIM + d0 + kv * 4;
        const float4 f = *(const float4*)src;
        dst[0] = (_Float16)f.x; dst[1] = (_Float16)f.y;
        dst[2] = (_Float16)f.z; dst[3] = (_Float16)f.w;
        if (kc + 1 < NKC) __builtin_prefetch(src + KC, 0, 1);  // global_prefetch_b8
      } else {
        dst[0] = (_Float16)0.f; dst[1] = (_Float16)0.f;
        dst[2] = (_Float16)0.f; dst[3] = (_Float16)0.f;
      }
    }
    __syncthreads();

    // ---- compute: 2 WMMA K-steps of 32 over this chunk ----
#pragma unroll
    for (int ks = 0; ks < KC; ks += 32) {
      v16h afrag[2], bfrag[2];
      // A 16x32 f16 layout: lanes 0-15 halfs = K[0..7],K[16..23]; lanes 16-31 = K[8..15],K[24..31]
#pragma unroll
      for (int m = 0; m < 2; ++m) {
        const _Float16* ap = &As[(m * 16 + lane16) * A_STR + ks + laneHi * 8];
        const v8h lo = *(const v8h*)ap;
        const v8h hi = *(const v8h*)(ap + 16);
        afrag[m] = __builtin_shufflevector(lo, hi, 0,1,2,3,4,5,6,7,8,9,10,11,12,13,14,15);
      }
      // B 32x16 f16 layout: lane holds one column; lanes 0-15 = K[0..7],K[16..23]; lanes 16-31 = K[8..15],K[24..31]
#pragma unroll
      for (int n = 0; n < 2; ++n) {
        const int col = (wave * 2 + n) * 16 + lane16;
        const _Float16* bp = &Bs[col * B_STR + ks + laneHi * 8];
        const v8h lo = *(const v8h*)bp;
        const v8h hi = *(const v8h*)(bp + 16);
        bfrag[n] = __builtin_shufflevector(lo, hi, 0,1,2,3,4,5,6,7,8,9,10,11,12,13,14,15);
      }
#pragma unroll
      for (int m = 0; m < 2; ++m)
#pragma unroll
        for (int n = 0; n < 2; ++n)
          acc[m][n] = __builtin_amdgcn_wmma_f32_16x16x32_f16(
              false, afrag[m], false, bfrag[n], (short)0, acc[m][n], false, false);
    }
  }

  // ---- row-max over N (zero-padded cols make this relu(max) exactly) ----
  // C layout: VGPR r: lanes 0-15 -> (M=r, N=lane16), lanes 16-31 -> (M=8+r, N=lane16)
  float rmax[2][8];
#pragma unroll
  for (int m = 0; m < 2; ++m)
#pragma unroll
    for (int r = 0; r < 8; ++r) {
      float v = fmaxf(acc[m][0][r], acc[m][1][r]);   // max over this wave's 2 Ntiles
#pragma unroll
      for (int mask = 1; mask <= 8; mask <<= 1)      // max across 16 lane-columns
        v = fmaxf(v, __shfl_xor(v, mask, 32));
      rmax[m][r] = v;
    }
  if (lane16 == 0) {
#pragma unroll
    for (int m = 0; m < 2; ++m)
#pragma unroll
      for (int r = 0; r < 8; ++r)
        WMax[wave * 32 + m * 16 + laneHi * 8 + r] = rmax[m][r];
  }
  __syncthreads();

  // ---- cross-wave max, log1p, sum over 32 query tokens ----
  if (wave == 0) {
    float v = WMax[lane];
    for (int w = 1; w < NWAVES; ++w) v = fmaxf(v, WMax[w * 32 + lane]);
    float s = log1pf(v);
#pragma unroll
    for (int mask = 16; mask >= 1; mask >>= 1) s += __shfl_xor(s, mask, 32);
    if (lane == 0) Out[q * P_NUM + p] = s;
  }
}

extern "C" void kernel_launch(void* const* d_in, const int* in_sizes, int n_in,
                              void* d_out, int out_size, void* d_ws, size_t ws_size,
                              hipStream_t stream) {
  (void)in_sizes; (void)n_in; (void)d_ws; (void)ws_size; (void)out_size;
  const float* Qe = (const float*)d_in[0];   // [32, 32, 768] fp32
  const float* Pe = (const float*)d_in[1];   // [256, 300, 768] fp32
  float* Out = (float*)d_out;                // [32, 256] fp32
  dim3 grid(Q_NUM, P_NUM);                   // x = query (fast) => blocks sharing p are adjacent
  dim3 block(NTHREADS);
  SPARTA_colbert_wmma<<<grid, block, 0, stream>>>(Qe, Pe, Out);
}
